// MambaBackbone_73581379715115
// MI455X (gfx1250) — matrix operand order repaired and checked by reading, not compiled, not run on hardware
//
#include <hip/hip_runtime.h>
#include <hip/hip_bf16.h>

// ---------------- model dims ----------------
#define B_   2
#define L_   1024
#define DM   512          // d_model
#define DI   1024         // d_inner
#define DS_  16           // d_state
#define DTR  32           // dt_rank
#define NROW (B_ * L_)    // 2048 token rows

typedef __bf16 bf16_t;
typedef __attribute__((ext_vector_type(16))) __bf16 v16bf;
typedef __attribute__((ext_vector_type(8)))  __bf16 v8bf;   // 16B = one b128 load
typedef __attribute__((ext_vector_type(8)))  float  v8f;

union frag16 { v16bf v; v8bf h[2]; };

// round-to-nearest-even f32 -> bf16
__device__ __forceinline__ bf16_t f2bf(float f) {
    unsigned u = __builtin_bit_cast(unsigned, f);
    unsigned r = (u + 0x7FFFu + ((u >> 16) & 1u)) >> 16;
    unsigned short s = (unsigned short)r;
    return __builtin_bit_cast(bf16_t, s);
}
__device__ __forceinline__ bf16_t bf_zero() {
    unsigned short z = 0;
    return __builtin_bit_cast(bf16_t, z);
}

// =====================================================================
// f32 -> bf16 bulk conversion (weights, once per launch)
// =====================================================================
__global__ void convert_bf16_kernel(const float* __restrict__ src,
                                    bf16_t* __restrict__ dst, int n)
{
    int i = blockIdx.x * blockDim.x + threadIdx.x;
    if (i < n) dst[i] = f2bf(src[i]);
}

// dt_proj weights: [8][DI][32] f32 -> [8][DI][64] bf16, zero-padded K 32..63
// so the dt GEMM can run the K%64==0 branchless pipeline against the full
// 64-wide x_dbl rows (padded weights zero out the B/C columns).
__global__ void convert_pad_dtw_kernel(const float* __restrict__ src,
                                       bf16_t* __restrict__ dst, int n)
{
    int i = blockIdx.x * blockDim.x + threadIdx.x;
    if (i >= n) return;
    int col  = i & 63;
    int rowp = i >> 6;
    dst[i] = (col < DTR) ? f2bf(src[rowp * DTR + col]) : bf_zero();
}

// =====================================================================
// GEMM: C[M,N] = A[M,K] * W[N,K]^T   (optionally += ), K % 64 == 0.
// A, W in bf16 memory; fragments load as raw b128s in WMMA layout.
// Wave computes a 16(M) x 64(N) strip; 8 waves / block stacked along M.
// Branchless software pipeline: ping-pong fragment sets with clamped
// (dummy) prefetch on the final stage -> single-basic-block inner loop,
// enabling partial s_wait_loadcnt and real load/WMMA overlap.
// =====================================================================
__global__ __launch_bounds__(256)
void gemm_wmma(const bf16_t* __restrict__ A, int lda,
               const bf16_t* __restrict__ W,   // [N x K] row-major bf16
               float* __restrict__ C, int ldc,
               bf16_t* __restrict__ Cbf,       // optional bf16 mirror of C
               int M, int N, int K, int accumulate)
{
    const int lane = threadIdx.x & 31;
    const int wave = threadIdx.x >> 5;
    const int m0 = blockIdx.y * 128 + wave * 16;
    const int n0 = blockIdx.x * 64;
    if (m0 >= M) return;

    const int mn = lane & 15;   // m for A / n for B,C,D
    const int kh = lane >> 4;   // K-half selector / M-half for C,D

    v8f acc[4];
    if (accumulate) {
        #pragma unroll
        for (int nt = 0; nt < 4; ++nt)
            #pragma unroll
            for (int r = 0; r < 8; ++r)
                acc[nt][r] = C[(long)(m0 + kh * 8 + r) * ldc + n0 + nt * 16 + mn];
    } else {
        #pragma unroll
        for (int nt = 0; nt < 4; ++nt)
            #pragma unroll
            for (int r = 0; r < 8; ++r)
                acc[nt][r] = 0.0f;
    }

    // A 16x32 bf16 layout: lane m=mn; kh=0 -> K 0-7 in V0-3, 16-23 in V4-7;
    //                      kh=1 -> K 8-15 in V0-3, 24-31 in V4-7.
    const bf16_t* arow  = A + (long)(m0 + mn) * lda + 8 * kh;
    // B 32x16 bf16 layout: lane n=mn; same K order as A (kh=0 -> K 0-7,16-23; kh=1 -> K 8-15,24-31).
    const bf16_t* wbase = W + (long)(n0 + mn) * K + 8 * kh;
    const long wstride = (long)16 * K;

    frag16 afA, bfA[4], afB, bfB[4];

    // prologue: stage at kk = 0
    afA.h[0] = *(const v8bf*)(arow);
    afA.h[1] = *(const v8bf*)(arow + 16);
    #pragma unroll
    for (int nt = 0; nt < 4; ++nt) {
        bfA[nt].h[0] = *(const v8bf*)(wbase + nt * wstride);
        bfA[nt].h[1] = *(const v8bf*)(wbase + nt * wstride + 16);
    }

    for (int kk = 0; kk < K; kk += 64) {
        // prefetch odd stage (always real: K % 64 == 0 -> kk+32 < K)
        const int k1 = kk + 32;
        afB.h[0] = *(const v8bf*)(arow + k1);
        afB.h[1] = *(const v8bf*)(arow + k1 + 16);
        #pragma unroll
        for (int nt = 0; nt < 4; ++nt) {
            bfB[nt].h[0] = *(const v8bf*)(wbase + nt * wstride + k1);
            bfB[nt].h[1] = *(const v8bf*)(wbase + nt * wstride + k1 + 16);
        }
        #pragma unroll
        for (int nt = 0; nt < 4; ++nt)
            acc[nt] = __builtin_amdgcn_wmma_f32_16x16x32_bf16(
                false, afA.v, false, bfA[nt].v, (short)0, acc[nt], false, false);

        // prefetch next even stage; clamp to 0 on last iteration (dummy
        // load, never consumed) to keep the loop a single basic block.
        const int k2n = kk + 64;
        const int k2  = (k2n < K) ? k2n : 0;
        afA.h[0] = *(const v8bf*)(arow + k2);
        afA.h[1] = *(const v8bf*)(arow + k2 + 16);
        #pragma unroll
        for (int nt = 0; nt < 4; ++nt) {
            bfA[nt].h[0] = *(const v8bf*)(wbase + nt * wstride + k2);
            bfA[nt].h[1] = *(const v8bf*)(wbase + nt * wstride + k2 + 16);
        }
        #pragma unroll
        for (int nt = 0; nt < 4; ++nt)
            acc[nt] = __builtin_amdgcn_wmma_f32_16x16x32_bf16(
                false, afB.v, false, bfB[nt].v, (short)0, acc[nt], false, false);
    }

    #pragma unroll
    for (int nt = 0; nt < 4; ++nt)
        #pragma unroll
        for (int r = 0; r < 8; ++r) {
            long idx = (long)(m0 + kh * 8 + r) * ldc + n0 + nt * 16 + mn;
            C[idx] = acc[nt][r];
            if (Cbf) Cbf[idx] = f2bf(acc[nt][r]);
        }
}

// =====================================================================
// LayerNorm over D=512; one block (256 threads) per token row.
// bf16 outputs (straight + time-flipped) feeding the in_proj GEMMs.
// =====================================================================
__global__ __launch_bounds__(256)
void layernorm_bf16_kernel(const float* __restrict__ x,
                           const float* __restrict__ w, const float* __restrict__ b,
                           bf16_t* __restrict__ out, bf16_t* __restrict__ out_flip)
{
    __shared__ float red[256];
    const int row = blockIdx.x;
    const int tid = threadIdx.x;
    const float* xr = x + (long)row * DM;

    float v0 = xr[tid], v1 = xr[tid + 256];
    red[tid] = v0 + v1;
    __syncthreads();
    for (int o = 128; o > 0; o >>= 1) { if (tid < o) red[tid] += red[tid + o]; __syncthreads(); }
    float mean = red[0] * (1.0f / DM);
    __syncthreads();

    float d0 = v0 - mean, d1 = v1 - mean;
    red[tid] = d0 * d0 + d1 * d1;
    __syncthreads();
    for (int o = 128; o > 0; o >>= 1) { if (tid < o) red[tid] += red[tid + o]; __syncthreads(); }
    float rstd = rsqrtf(red[0] * (1.0f / DM) + 1e-5f);

    float o0 = d0 * rstd * w[tid]       + b[tid];
    float o1 = d1 * rstd * w[tid + 256] + b[tid + 256];
    out[(long)row * DM + tid]       = f2bf(o0);
    out[(long)row * DM + tid + 256] = f2bf(o1);
    int bb = row / L_, ll = row % L_;
    long fr = (long)(bb * L_ + (L_ - 1 - ll)) * DM;
    out_flip[fr + tid]       = f2bf(o0);
    out_flip[fr + tid + 256] = f2bf(o1);
}

// Final LayerNorm: f32 output to d_out.
__global__ __launch_bounds__(256)
void layernorm_f32_kernel(const float* __restrict__ x,
                          const float* __restrict__ w, const float* __restrict__ b,
                          float* __restrict__ out)
{
    __shared__ float red[256];
    const int row = blockIdx.x;
    const int tid = threadIdx.x;
    const float* xr = x + (long)row * DM;

    float v0 = xr[tid], v1 = xr[tid + 256];
    red[tid] = v0 + v1;
    __syncthreads();
    for (int o = 128; o > 0; o >>= 1) { if (tid < o) red[tid] += red[tid + o]; __syncthreads(); }
    float mean = red[0] * (1.0f / DM);
    __syncthreads();

    float d0 = v0 - mean, d1 = v1 - mean;
    red[tid] = d0 * d0 + d1 * d1;
    __syncthreads();
    for (int o = 128; o > 0; o >>= 1) { if (tid < o) red[tid] += red[tid + o]; __syncthreads(); }
    float rstd = rsqrtf(red[0] * (1.0f / DM) + 1e-5f);

    out[(long)row * DM + tid]       = d0 * rstd * w[tid]       + b[tid];
    out[(long)row * DM + tid + 256] = d1 * rstd * w[tid + 256] + b[tid + 256];
}

// =====================================================================
// Depthwise causal conv (k=4) + bias + SiLU; emits f32 (scan) + bf16 (GEMM).
// =====================================================================
__global__ void conv_silu_kernel(const float* __restrict__ xz,   // [NROW][2*DI]
                                 const float* __restrict__ cw,   // [DI][4]
                                 const float* __restrict__ cb,   // [DI]
                                 float* __restrict__ xc,         // [NROW][DI]
                                 bf16_t* __restrict__ xc_bf)     // [NROW][DI]
{
    int idx = blockIdx.x * blockDim.x + threadIdx.x;
    if (idx >= NROW * DI) return;
    int d   = idx & (DI - 1);
    int row = idx >> 10;           // DI == 1024
    int l   = row & (L_ - 1);      // L_ == 1024
    float acc = cb[d];
    #pragma unroll
    for (int k = 0; k < 4; ++k) {
        int t = l - 3 + k;
        if (t >= 0) acc += xz[(long)(row - l + t) * (2 * DI) + d] * cw[d * 4 + k];
    }
    float s = acc / (1.0f + __expf(-acc));
    xc[idx]    = s;
    xc_bf[idx] = f2bf(s);
}

// =====================================================================
// dt = softplus(dt + bias)   (in place, f32)
// =====================================================================
__global__ void softplus_bias_kernel(float* __restrict__ dt,
                                     const float* __restrict__ bias)
{
    int idx = blockIdx.x * blockDim.x + threadIdx.x;
    if (idx >= NROW * DI) return;
    float v = dt[idx] + bias[idx & (DI - 1)];
    dt[idx] = (v > 20.0f) ? v : log1pf(__expf(v));
}

// =====================================================================
// Selective scan fused with +u*D and *silu(z); bf16 gated output
// (its only consumer is the out_proj WMMA GEMM).
// One thread per (batch, channel); B_t/C_t broadcast via double-buffered LDS.
// =====================================================================
__global__ __launch_bounds__(256)
void scan_kernel(const float* __restrict__ u,     // xc    [NROW][DI]
                 const float* __restrict__ dt,    //       [NROW][DI]
                 const float* __restrict__ xdbl,  //       [NROW][64]  (B@32, C@48)
                 const float* __restrict__ xz,    //       [NROW][2*DI] (z at +DI)
                 const float* __restrict__ A_log, //       [DI][16]
                 const float* __restrict__ Dp,    //       [DI]
                 bf16_t* __restrict__ yg)         //       [NROW][DI] bf16
{
    __shared__ float sBC[2][32];
    const int bb = blockIdx.x / (DI / 256);
    const int d  = (blockIdx.x % (DI / 256)) * 256 + threadIdx.x;

    float A[16], h[16];
    #pragma unroll
    for (int n = 0; n < 16; ++n) {
        A[n] = -__expf(A_log[d * 16 + n]);
        h[n] = 0.0f;
    }
    const float Dd = Dp[d];

    for (int t = 0; t < L_; ++t) {
        const long row = (long)bb * L_ + t;
        if (threadIdx.x < 32)
            sBC[t & 1][threadIdx.x] = xdbl[row * 64 + 32 + threadIdx.x];
        __syncthreads();

        const float dtv = dt[row * DI + d];
        const float uv  = u [row * DI + d];
        const float dtu = dtv * uv;
        const float* sB = sBC[t & 1];
        const float* sC = sBC[t & 1] + 16;

        float y = 0.0f;
        #pragma unroll
        for (int n = 0; n < 16; ++n) {
            float dA = __expf(dtv * A[n]);
            h[n] = dA * h[n] + dtu * sB[n];
            y += h[n] * sC[n];
        }
        float z = xz[row * (2 * DI) + DI + d];
        float g = z / (1.0f + __expf(-z));
        yg[row * DI + d] = f2bf((y + uv * Dd) * g);
    }
}

// =====================================================================
// dst += src
// =====================================================================
__global__ void add_kernel(float* __restrict__ dst, const float* __restrict__ src, int n)
{
    int i = blockIdx.x * blockDim.x + threadIdx.x;
    if (i < n) dst[i] += src[i];
}

// =====================================================================
// host launcher
// =====================================================================
extern "C" void kernel_launch(void* const* d_in, const int* in_sizes, int n_in,
                              void* d_out, int out_size, void* d_ws, size_t ws_size,
                              hipStream_t stream)
{
    const float* x_in   = (const float*)d_in[0];
    const float* in_w   = (const float*)d_in[1];   // [4][2][2048][512]
    const float* conv_w = (const float*)d_in[2];   // [4][2][1024][4]
    const float* conv_b = (const float*)d_in[3];   // [4][2][1024]
    const float* xp_w   = (const float*)d_in[4];   // [4][2][64][1024]
    const float* dtp_w  = (const float*)d_in[5];   // [4][2][1024][32]
    const float* dtp_b  = (const float*)d_in[6];   // [4][2][1024]
    const float* A_log  = (const float*)d_in[7];   // [4][2][1024][16]
    const float* D_par  = (const float*)d_in[8];   // [4][2][1024]
    const float* out_w  = (const float*)d_in[9];   // [4][2][512][1024]
    const float* ln_w   = (const float*)d_in[10];  // [4][512]
    const float* ln_b   = (const float*)d_in[11];  // [4][512]
    const float* fn_w   = (const float*)d_in[12];  // [512]
    const float* fn_b   = (const float*)d_in[13];  // [512]

    const long NR = NROW;

    // ---- workspace layout: f32 region then bf16 region ----
    float* ws    = (float*)d_ws;
    float* cur   = ws;                    // [NR][DM]
    float* nxt   = cur   + NR * DM;       // [NR][DM]  (h1+h2 accumulator)
    float* resid = nxt   + NR * DM;       // [NR][DM]
    float* xz    = resid + NR * DM;       // [NR][2*DI]
    float* xc    = xz    + NR * 2 * DI;   // [NR][DI]
    float* dtb   = xc    + NR * DI;       // [NR][DI]
    float* xdb   = dtb   + NR * DI;       // [NR][64]

    bf16_t* bws    = (bf16_t*)(xdb + NR * 64);
    bf16_t* xn_bf  = bws;                        // [NR][DM]
    bf16_t* xnf_bf = xn_bf  + NR * DM;           // [NR][DM]
    bf16_t* xc_bf  = xnf_bf + NR * DM;           // [NR][DI]
    bf16_t* xdb_bf = xc_bf  + NR * DI;           // [NR][64]
    bf16_t* ygt_bf = xdb_bf + NR * 64;           // [NR][DI]
    bf16_t* in_wb  = ygt_bf + NR * DI;           // 8*2048*512
    bf16_t* xp_wb  = in_wb  + 8L * 2 * DI * DM;  // 8*64*1024
    bf16_t* dtp_wb = xp_wb  + 8L * 64 * DI;      // 8*1024*64 (zero-padded K)
    bf16_t* out_wb = dtp_wb + 8L * DI * 64;      // 8*512*1024

    const dim3 blk(256);

    // ---- one-time (per launch) weight conversion to bf16 ----
    {
        int n1 = 8 * 2 * DI * DM;  convert_bf16_kernel<<<(n1 + 255) / 256, blk, 0, stream>>>(in_w,  in_wb,  n1);
        int n2 = 8 * 64 * DI;      convert_bf16_kernel<<<(n2 + 255) / 256, blk, 0, stream>>>(xp_w,  xp_wb,  n2);
        int n3 = 8 * DI * 64;      convert_pad_dtw_kernel<<<(n3 + 255) / 256, blk, 0, stream>>>(dtp_w, dtp_wb, n3);
        int n4 = 8 * DM * DI;      convert_bf16_kernel<<<(n4 + 255) / 256, blk, 0, stream>>>(out_w, out_wb, n4);
    }

    hipMemcpyAsync(cur, x_in, NR * DM * sizeof(float), hipMemcpyDeviceToDevice, stream);

    for (int i = 0; i < 4; ++i) {
        if (i == 0)
            hipMemcpyAsync(resid, cur, NR * DM * sizeof(float),
                           hipMemcpyDeviceToDevice, stream);
        else
            add_kernel<<<(NR * DM) / 256, blk, 0, stream>>>(resid, cur, NR * DM);

        layernorm_bf16_kernel<<<NR, blk, 0, stream>>>(cur, ln_w + i * DM, ln_b + i * DM,
                                                      xn_bf, xnf_bf);

        for (int j = 0; j < 2; ++j) {
            const int p = i * 2 + j;
            const bf16_t* Asrc = j ? xnf_bf : xn_bf;

            // xz = A @ in_w^T            [2048 x 2048], K=512
            gemm_wmma<<<dim3(2 * DI / 64, NR / 128), blk, 0, stream>>>(
                Asrc, DM, in_wb + (long)p * 2 * DI * DM, xz, 2 * DI, nullptr,
                NR, 2 * DI, DM, 0);

            // xc = silu(causal depthwise conv(xc_half) + bias)  (f32 + bf16)
            conv_silu_kernel<<<(NR * DI) / 256, blk, 0, stream>>>(
                xz, conv_w + (long)p * DI * 4, conv_b + (long)p * DI, xc, xc_bf);

            // x_dbl = xc @ xp_w^T        [2048 x 64], K=1024  (f32 for scan + bf16 mirror)
            gemm_wmma<<<dim3(1, NR / 128), blk, 0, stream>>>(
                xc_bf, DI, xp_wb + (long)p * 64 * DI, xdb, 64, xdb_bf,
                NR, 64, DI, 0);

            // dt = x_dbl(64-wide) @ dtp_w_padded^T   [2048 x 1024], K=64
            gemm_wmma<<<dim3(DI / 64, NR / 128), blk, 0, stream>>>(
                xdb_bf, 64, dtp_wb + (long)p * DI * 64, dtb, DI, nullptr,
                NR, DI, 64, 0);
            softplus_bias_kernel<<<(NR * DI) / 256, blk, 0, stream>>>(
                dtb, dtp_b + (long)p * DI);

            // selective scan + *silu(z) fusion -> bf16 gated activations
            scan_kernel<<<B_ * (DI / 256), blk, 0, stream>>>(
                xc, dtb, xdb, xz, A_log + (long)p * DI * DS_, D_par + (long)p * DI, ygt_bf);

            // nxt (+)= ygated @ out_w^T  [2048 x 512], K=1024 ; j==1 accumulates h1+h2
            gemm_wmma<<<dim3(DM / 64, NR / 128), blk, 0, stream>>>(
                ygt_bf, DI, out_wb + (long)p * DM * DI, nxt, DM, nullptr,
                NR, DM, DI, j);
        }
        float* t = cur; cur = nxt; nxt = t;
    }

    add_kernel<<<(NR * DM) / 256, blk, 0, stream>>>(cur, resid, NR * DM);
    layernorm_f32_kernel<<<NR, blk, 0, stream>>>(cur, fn_w, fn_b, (float*)d_out);
}
